// DGCNN_4355096838557
// MI455X (gfx1250) — matrix operand order repaired and checked by reading, not compiled, not run on hardware
//
#include <hip/hip_runtime.h>
#include <hip/hip_bf16.h>

#define BATCH 4
#define NPTS  2048
#define KNN   20

typedef __attribute__((ext_vector_type(2)))  float    v2f;
typedef __attribute__((ext_vector_type(8)))  float    v8f;
typedef __attribute__((ext_vector_type(16))) _Float16 v16h;

// ---------------------------------------------------------------------------
// x (B,3,N) channel-major -> x0pm (B,N,4) point-major, zero 4th channel
// ---------------------------------------------------------------------------
__global__ void padx_kernel(const float* __restrict__ x, float* __restrict__ xpm) {
    int i = blockIdx.x * blockDim.x + threadIdx.x;      // B*N
    if (i >= BATCH * NPTS) return;
    int b = i / NPTS, n = i % NPTS;
    const float* xb = x + (size_t)b * 3 * NPTS;
    float4 v = make_float4(xb[n], xb[NPTS + n], xb[2 * NPTS + n], 0.f);
    ((float4*)xpm)[i] = v;
}

// pad W1 (64x6: [d0 d1 d2 c0 c1 c2]) -> W1p (64x8: [d0 d1 d2 0 c0 c1 c2 0])
__global__ void padw1_kernel(const float* __restrict__ W1, float* __restrict__ W1p) {
    int i = blockIdx.x * blockDim.x + threadIdx.x;      // 64*8
    if (i >= 64 * 8) return;
    int o = i >> 3, c = i & 7;
    int grp = c >> 2, cc = c & 3;
    W1p[i] = (cc < 3) ? W1[o * 6 + grp * 3 + cc] : 0.f;
}

// ---------------------------------------------------------------------------
// squared norms per point (point-major input -> contiguous float4 row sums)
// ---------------------------------------------------------------------------
template <int C>
__global__ void sqnorm_kernel(const float* __restrict__ xpm, float* __restrict__ sqn) {
    int i = blockIdx.x * blockDim.x + threadIdx.x;      // B*N
    if (i >= BATCH * NPTS) return;
    const float4* r = (const float4*)(xpm + (size_t)i * C);
    float s = 0.f;
#pragma unroll
    for (int q = 0; q < C / 4; ++q) {
        float4 v = r[q];
        s += v.x * v.x + v.y * v.y + v.z * v.z + v.w * v.w;
    }
    sqn[i] = s;
}

// ---------------------------------------------------------------------------
// negative squared distances via f32 WMMA 16x16x4 (point-major input).
// Block = 8 waves sharing one A tile (16 query rows, staged in LDS).
// ---------------------------------------------------------------------------
template <int C>
__global__ void dist_kernel(const float* __restrict__ xpm, const float* __restrict__ sqn,
                            float* __restrict__ D) {
    __shared__ float aT[C * 16];                      // [c][m] for rows i0..i0+15
    const int tjg_n = NPTS / 16 / 8;
    int blk = blockIdx.x;
    int b   = blk / ((NPTS / 16) * tjg_n);
    int r   = blk % ((NPTS / 16) * tjg_n);
    int ti  = r / tjg_n, tjg = r % tjg_n;
    int i0  = ti * 16;
    int tid = threadIdx.x;
    const float* xb = xpm + (size_t)b * NPTS * C;

    for (int f = tid; f < 16 * C; f += 256) {         // coalesced point-row reads
        int m = f / C, c = f % C;
        aT[c * 16 + m] = xb[(size_t)(i0 + m) * C + c];
    }
    __syncthreads();

    int lane = tid & 31, wave = tid >> 5;
    int l = lane & 15, hi = lane >> 4;
    int j0 = (tjg * 8 + wave) * 16;
    const float* brow = xb + (size_t)(j0 + l) * C;

    v8f acc = {};
#pragma unroll
    for (int s = 0; s < C / 4; ++s) {
        int ca = 4 * s + 2 * hi;                      // lanes0-15 K={0,1}, 16-31 K={2,3}
        v2f a, bb;
        a[0] = aT[ca * 16 + l];
        a[1] = aT[(ca + 1) * 16 + l];
        bb   = *(const v2f*)(brow + ca);              // one b64 per step
        acc = __builtin_amdgcn_wmma_f32_16x16x4_f32(false, a, false, bb,
                                                    (short)0, acc, false, false);
    }
    const float* sq = sqn + b * NPTS;
    float sj = sq[j0 + l];
    float* Db = D + (size_t)b * NPTS * NPTS;
#pragma unroll
    for (int v = 0; v < 8; ++v) {
        int m = v + 8 * hi;
        float nd = 2.f * acc[v] - sq[i0 + m] - sj;
        Db[(size_t)(i0 + m) * NPTS + j0 + l] = nd;
    }
}

// ---------------------------------------------------------------------------
// top-K per row, lax.top_k semantics (descending, ties -> lower index)
// ---------------------------------------------------------------------------
__global__ void topk_kernel(const float* __restrict__ D, int* __restrict__ knn) {
    __shared__ float vals[NPTS];
    __shared__ float wv[8];
    __shared__ int   wi[8];
    int row = blockIdx.x, tid = threadIdx.x;
    const float4* Dr4 = (const float4*)(D + (size_t)row * NPTS);
    float4* v4 = (float4*)vals;
    for (int j = tid; j < NPTS / 4; j += 256) v4[j] = Dr4[j];
    __syncthreads();
    int lane = tid & 31, wave = tid >> 5;
    for (int t = 0; t < KNN; ++t) {
        float best = -__builtin_inff(); int bi = NPTS;
        for (int j = tid; j < NPTS; j += 256) {
            float v = vals[j];
            if (v > best) { best = v; bi = j; }
        }
#pragma unroll
        for (int off = 16; off > 0; off >>= 1) {
            float ov = __shfl_xor(best, off, 32);
            int   oi = __shfl_xor(bi, off, 32);
            if (ov > best || (ov == best && oi < bi)) { best = ov; bi = oi; }
        }
        if (lane == 0) { wv[wave] = best; wi[wave] = bi; }
        __syncthreads();
        if (tid == 0) {
            float gb = wv[0]; int gi = wi[0];
#pragma unroll
            for (int u = 1; u < 8; ++u)
                if (wv[u] > gb || (wv[u] == gb && wi[u] < gi)) { gb = wv[u]; gi = wi[u]; }
            knn[(size_t)row * KNN + t] = gi;
            vals[gi] = -__builtin_inff();
        }
        __syncthreads();
    }
}

// ---------------------------------------------------------------------------
// fused edge-conv (point-major activations):
//   gather(K=20) + (O x 2C) GEMM (f16 WMMA, f32 acc) + BN + LReLU + max over K
// 8 waves/block share a 16-row weight tile (LDS, pre-swizzled); k-invariant
// operands hoisted: center-half B fragments (bPre) and the center term of the
// diff-half (ctrPre, f32) live in registers; per neighbor only 4 b128 loads
// per diff chunk remain.
// ---------------------------------------------------------------------------
template <int C, int O>
__global__ void edgeconv_kernel(const float* __restrict__ xpm, const float* __restrict__ W,
                                const float* __restrict__ g, const float* __restrict__ beta,
                                const int* __restrict__ nbr, float* __restrict__ xopm) {
    const int C2 = 2 * C;
    const int STEPS = (C2 + 31) / 32;
    __shared__ __align__(32) _Float16 aLds[STEPS * 32 * 16];
    const int otiles = O / 16, ngroups = NPTS / 16 / 8;
    int blk = blockIdx.x;
    int b  = blk / (otiles * ngroups);
    int r  = blk % (otiles * ngroups);
    int ot = r / ngroups, ng = r % ngroups;
    int o0 = ot * 16;
    int tid = threadIdx.x;

    // stage A fragments (weight rows o0..o0+15, all K-steps), pre-swizzled
    for (int e = tid; e < STEPS * 32; e += 256) {
        int s = e >> 5, ln = e & 31;
        int ll = ln & 15, hh = ln >> 4;
        const float* Wr = W + (size_t)(o0 + ll) * C2;
        _Float16* dst = &aLds[(size_t)e * 16];
#pragma unroll
        for (int v = 0; v < 8; ++v) {
            int k0  = 2 * v + 16 * hh;
            int cc0 = s * 32 + k0;
            float w0 = Wr[(cc0 < C2) ? cc0 : 0];
            float w1 = Wr[(cc0 + 1 < C2) ? cc0 + 1 : 0];
            dst[2 * v]     = (cc0 < C2)     ? (_Float16)w0 : (_Float16)0.f;
            dst[2 * v + 1] = (cc0 + 1 < C2) ? (_Float16)w1 : (_Float16)0.f;
        }
    }
    __syncthreads();

    int wave = tid >> 5, lane = tid & 31;
    int l = lane & 15, hi = lane >> 4;
    int n0 = (ng * 8 + wave) * 16;
    int p  = n0 + l;
    const float* xb = xpm + (size_t)b * NPTS * C;
    const int* nb_row = nbr + ((size_t)b * NPTS + p) * KNN;
    const v16h* aV = (const v16h*)aLds;

    v16h aReg[STEPS];
#pragma unroll
    for (int s = 0; s < STEPS; ++s) aReg[s] = aV[s * 32 + lane];

    // k-invariant precomputes
    float4 c4 = make_float4(0.f, 0.f, 0.f, 0.f);      // layer-1 center row
    float ctrPre[STEPS][16];                          // center term of diff chunks
    v16h  bPre[STEPS];                                // full center-half fragments
    if (C == 4) {
        c4 = *(const float4*)(xb + (size_t)p * 4);
    } else {
#pragma unroll
        for (int s = 0; s < STEPS; ++s) {
            if (s * 32 < C) {                          // pure diff chunk
                const float4* cp = (const float4*)(xb + (size_t)p * C + s * 32 + 16 * hi);
#pragma unroll
                for (int q = 0; q < 4; ++q) ((float4*)&ctrPre[s][0])[q] = cp[q];
            } else {                                   // pure center chunk
                const float4* cp = (const float4*)(xb + (size_t)p * C + (s * 32 - C) + 16 * hi);
                float t[16];
#pragma unroll
                for (int q = 0; q < 4; ++q) ((float4*)t)[q] = cp[q];
                v16h bv;
#pragma unroll
                for (int q = 0; q < 16; ++q) bv[q] = (_Float16)t[q];
                bPre[s] = bv;
            }
        }
    }

    const float rs = rsqrtf(1.f + 1e-5f);
    float sc[8], bs[8];
#pragma unroll
    for (int v = 0; v < 8; ++v) {
        int m = v + 8 * hi;
        sc[v] = g[o0 + m] * rs;
        bs[v] = beta[o0 + m];
    }

    v8f best;
#pragma unroll
    for (int v = 0; v < 8; ++v) best[v] = -__builtin_inff();

    for (int k = 0; k < KNN; ++k) {
        int nb = nb_row[k];
        v8f acc = {};
#pragma unroll
        for (int s = 0; s < STEPS; ++s) {
            v16h bv;
            if (C == 4) {
                // single mixed chunk: channels [diff0..3, ctr0..3, 0...]
                float4 n4 = *(const float4*)(xb + (size_t)nb * 4);
                float e[8] = { n4.x - c4.x, n4.y - c4.y, n4.z - c4.z, n4.w - c4.w,
                               c4.x, c4.y, c4.z, c4.w };
#pragma unroll
                for (int q = 0; q < 16; ++q)
                    bv[q] = (hi == 0 && q < 8) ? (_Float16)e[q < 8 ? q : 0] : (_Float16)0.f;
            } else if (s * 32 < C) {
                const float4* np4 = (const float4*)(xb + (size_t)nb * C + s * 32 + 16 * hi);
                float ngh[16];
#pragma unroll
                for (int q = 0; q < 4; ++q) ((float4*)ngh)[q] = np4[q];
#pragma unroll
                for (int q = 0; q < 16; ++q) bv[q] = (_Float16)(ngh[q] - ctrPre[s][q]);
            } else {
                bv = bPre[s];                          // neighbor-independent chunk
            }
            acc = __builtin_amdgcn_wmma_f32_16x16x32_f16(false, aReg[s], false, bv,
                                                         (short)0, acc, false, false);
        }
#pragma unroll
        for (int v = 0; v < 8; ++v) {                  // BN + LeakyReLU, then max over k
            float y = acc[v] * sc[v] + bs[v];
            y = (y >= 0.f) ? y : 0.2f * y;
            best[v] = fmaxf(best[v], y);
        }
    }
    // point-major output: this lane owns point p, channels o0+8*hi .. +7
    float* orow = xopm + ((size_t)b * NPTS + p) * O + o0 + 8 * hi;
    ((float4*)orow)[0] = make_float4(best[0], best[1], best[2], best[3]);
    ((float4*)orow)[1] = make_float4(best[4], best[5], best[6], best[7]);
}

// ---------------------------------------------------------------------------
// layer 5: (1024 x 256) conv + BN + LReLU + per-tile max over 16 columns
// ---------------------------------------------------------------------------
__global__ void conv5_kernel(const float* __restrict__ x4pm, const float* __restrict__ W,
                             const float* __restrict__ g, const float* __restrict__ beta,
                             float* __restrict__ part) {
    const int C = 256, O = 1024, STEPS = 8;
    __shared__ __align__(32) _Float16 aLds[STEPS * 32 * 16];
    const int otiles = O / 16, ngroups = NPTS / 16 / 8;
    int blk = blockIdx.x;
    int b  = blk / (otiles * ngroups);
    int r  = blk % (otiles * ngroups);
    int ot = r / ngroups, ng = r % ngroups;
    int o0 = ot * 16;
    int tid = threadIdx.x;

    for (int e = tid; e < STEPS * 32; e += 256) {
        int s = e >> 5, ln = e & 31;
        int ll = ln & 15, hh = ln >> 4;
        const float* Wr = W + (size_t)(o0 + ll) * C;
        _Float16* dst = &aLds[(size_t)e * 16];
#pragma unroll
        for (int v = 0; v < 8; ++v) {
            int k0  = 2 * v + 16 * hh;
            int cc0 = s * 32 + k0;
            dst[2 * v]     = (_Float16)Wr[cc0];
            dst[2 * v + 1] = (_Float16)Wr[cc0 + 1];
        }
    }
    __syncthreads();

    int wave = tid >> 5, lane = tid & 31, l = lane & 15, hi = lane >> 4;
    int nt = ng * 8 + wave, n0 = nt * 16;
    const float* xrow = x4pm + ((size_t)b * NPTS + n0 + l) * C;
    const v16h* aV = (const v16h*)aLds;

    v8f acc = {};
#pragma unroll
    for (int s = 0; s < STEPS; ++s) {
        v16h a = aV[s * 32 + lane];
        const float4* bp = (const float4*)(xrow + s * 32 + 16 * hi);
        float t[16];
#pragma unroll
        for (int q = 0; q < 4; ++q) ((float4*)t)[q] = bp[q];
        v16h bv;
#pragma unroll
        for (int q = 0; q < 16; ++q) bv[q] = (_Float16)t[q];
        acc = __builtin_amdgcn_wmma_f32_16x16x32_f16(false, a, false, bv,
                                                     (short)0, acc, false, false);
    }
    const float rs = rsqrtf(1.f + 1e-5f);
#pragma unroll
    for (int v = 0; v < 8; ++v) {
        int m = v + 8 * hi;
        float y = acc[v] * (g[o0 + m] * rs) + beta[o0 + m];
        y = (y >= 0.f) ? y : 0.2f * y;
#pragma unroll
        for (int off = 8; off > 0; off >>= 1)          // max over 16 cols per half-wave
            y = fmaxf(y, __shfl_xor(y, off, 32));
        if (l == 0)
            part[((size_t)b * O + (o0 + m)) * (NPTS / 16) + nt] = y;
    }
}

__global__ void reduce5_kernel(const float* __restrict__ part, float* __restrict__ out) {
    int i = blockIdx.x * blockDim.x + threadIdx.x;     // B*1024
    if (i >= BATCH * 1024) return;
    const float* p = part + (size_t)i * (NPTS / 16);
    float m = -__builtin_inff();
#pragma unroll 8
    for (int t = 0; t < NPTS / 16; ++t) m = fmaxf(m, p[t]);
    out[i] = m;
}

// ---------------------------------------------------------------------------
extern "C" void kernel_launch(void* const* d_in, const int* in_sizes, int n_in,
                              void* d_out, int out_size, void* d_ws, size_t ws_size,
                              hipStream_t stream) {
    (void)in_sizes; (void)n_in; (void)out_size; (void)ws_size;
    // setup_inputs() dict order: x, then (W,g,b) per layer
    const float* x  = (const float*)d_in[0];
    const float* W1 = (const float*)d_in[1];
    const float* g1 = (const float*)d_in[2];
    const float* b1 = (const float*)d_in[3];
    const float* W2 = (const float*)d_in[4];
    const float* g2 = (const float*)d_in[5];
    const float* b2 = (const float*)d_in[6];
    const float* W3 = (const float*)d_in[7];
    const float* g3 = (const float*)d_in[8];
    const float* b3 = (const float*)d_in[9];
    const float* W4 = (const float*)d_in[10];
    const float* g4 = (const float*)d_in[11];
    const float* b4 = (const float*)d_in[12];
    const float* W5 = (const float*)d_in[13];
    const float* g5 = (const float*)d_in[14];
    const float* b5 = (const float*)d_in[15];
    float* out = (float*)d_out;

    // workspace carve (~83MiB, resident in MI455X's 192MB L2)
    char* w = (char*)d_ws;
    float* sqn  = (float*)w;  w += (size_t)BATCH * NPTS * sizeof(float);
    float* D    = (float*)w;  w += (size_t)BATCH * NPTS * NPTS * sizeof(float);
    int*   idx  = (int*)w;    w += (size_t)BATCH * NPTS * KNN * sizeof(int);
    float* x0pm = (float*)w;  w += (size_t)BATCH * NPTS * 4   * sizeof(float);
    float* x1pm = (float*)w;  w += (size_t)BATCH * NPTS * 64  * sizeof(float);
    float* x2pm = (float*)w;  w += (size_t)BATCH * NPTS * 64  * sizeof(float);
    float* x3pm = (float*)w;  w += (size_t)BATCH * NPTS * 128 * sizeof(float);
    float* x4pm = (float*)w;  w += (size_t)BATCH * NPTS * 256 * sizeof(float);
    float* W1p  = (float*)w;  w += (size_t)64 * 8 * sizeof(float);
    float* p5   = (float*)w;  w += (size_t)BATCH * 1024 * (NPTS / 16) * sizeof(float);

    const int distBlocks = BATCH * (NPTS / 16) * (NPTS / 16 / 8);
    const int sqBlocks   = (BATCH * NPTS + 255) / 256;

    padx_kernel<<<sqBlocks, 256, 0, stream>>>(x, x0pm);
    padw1_kernel<<<2, 256, 0, stream>>>(W1, W1p);

    // ---- layer 1 (C=4 padded, O=64) ----
    sqnorm_kernel<4><<<sqBlocks, 256, 0, stream>>>(x0pm, sqn);
    dist_kernel<4><<<distBlocks, 256, 0, stream>>>(x0pm, sqn, D);
    topk_kernel<<<BATCH * NPTS, 256, 0, stream>>>(D, idx);
    edgeconv_kernel<4, 64><<<BATCH * (64 / 16) * 16, 256, 0, stream>>>(x0pm, W1p, g1, b1, idx, x1pm);

    // ---- layer 2 (C=64, O=64) ----
    sqnorm_kernel<64><<<sqBlocks, 256, 0, stream>>>(x1pm, sqn);
    dist_kernel<64><<<distBlocks, 256, 0, stream>>>(x1pm, sqn, D);
    topk_kernel<<<BATCH * NPTS, 256, 0, stream>>>(D, idx);
    edgeconv_kernel<64, 64><<<BATCH * (64 / 16) * 16, 256, 0, stream>>>(x1pm, W2, g2, b2, idx, x2pm);

    // ---- layer 3 (C=64, O=128) ----
    sqnorm_kernel<64><<<sqBlocks, 256, 0, stream>>>(x2pm, sqn);
    dist_kernel<64><<<distBlocks, 256, 0, stream>>>(x2pm, sqn, D);
    topk_kernel<<<BATCH * NPTS, 256, 0, stream>>>(D, idx);
    edgeconv_kernel<64, 128><<<BATCH * (128 / 16) * 16, 256, 0, stream>>>(x2pm, W3, g3, b3, idx, x3pm);

    // ---- layer 4 (C=128, O=256) ----
    sqnorm_kernel<128><<<sqBlocks, 256, 0, stream>>>(x3pm, sqn);
    dist_kernel<128><<<distBlocks, 256, 0, stream>>>(x3pm, sqn, D);
    topk_kernel<<<BATCH * NPTS, 256, 0, stream>>>(D, idx);
    edgeconv_kernel<128, 256><<<BATCH * (256 / 16) * 16, 256, 0, stream>>>(x3pm, W4, g4, b4, idx, x4pm);

    // ---- layer 5 + global max ----
    conv5_kernel<<<BATCH * (1024 / 16) * 16, 256, 0, stream>>>(x4pm, W5, g5, b5, p5);
    reduce5_kernel<<<(BATCH * 1024 + 255) / 256, 256, 0, stream>>>(p5, out);
}
